// SNN_16269336117866
// MI455X (gfx1250) — matrix operand order repaired and checked by reading, not compiled, not run on hardware
//
#include <hip/hip_runtime.h>

typedef __attribute__((ext_vector_type(16))) _Float16 v16h;
typedef __attribute__((ext_vector_type(8)))  _Float16 v8h;
typedef __attribute__((ext_vector_type(4)))  _Float16 v4h;
typedef __attribute__((ext_vector_type(8)))  float    v8f;
typedef __attribute__((ext_vector_type(4)))  float    v4f;

#define T_STEPS 64
#define BATCH   256
#define K_IN    2048     // N_IN
#define N_TOT   2048     // N1 + N2 (W1 || W2 concatenated along output dim)
#define N_LAYER 1024
#define TBN     ((size_t)T_STEPS * BATCH * N_LAYER)   // elems per output array

union FragAB { v16h v; v8h h[2]; };

// WGP-scope (scope 0) prefetch: pulls the line into ALL cache levels,
// including the WGP cache. The __builtin_prefetch path can only emit
// SE/SYS scope on gfx1250, which stops at GL2 -- useless for data that is
// already L2-resident. Inline asm encodes the near-scope form directly.
__device__ __forceinline__ void prefetch_wgp(const void* p) {
    asm volatile("global_prefetch_b8 %0, off" :: "v"(p));
}

// ---------------------------------------------------------------------------
// f32 -> f16 conversion of the spike train (values are exactly 0.0 / 1.0).
// Source is single-use -> non-temporal loads keep it from evicting the
// L2-resident working set; the f16 destination IS the working set -> RT store.
// ---------------------------------------------------------------------------
__global__ __launch_bounds__(256) void cvt_spikes_f16(const float* __restrict__ src,
                                                      _Float16* __restrict__ dst,
                                                      int n4) {
    int i = blockIdx.x * blockDim.x + threadIdx.x;
    if (i >= n4) return;
    v4f f = __builtin_nontemporal_load((const v4f* __restrict__)src + i);
    v4h h;
    h[0] = (_Float16)f[0];
    h[1] = (_Float16)f[1];
    h[2] = (_Float16)f[2];
    h[3] = (_Float16)f[3];
    ((v4h* __restrict__)dst)[i] = h;
}

// ---------------------------------------------------------------------------
// Pack W1 (rows 0..1023) and W2 (rows 1024..2047) into one [2048][2048] f16
// row-major matrix. Row-major [n][k] means a WMMA B-matrix *column* is a
// contiguous memory row -> no transpose loads needed.
// ---------------------------------------------------------------------------
__global__ __launch_bounds__(256) void pack_weights_f16(const float* __restrict__ W1,
                                                        const float* __restrict__ W2,
                                                        _Float16* __restrict__ w16) {
    int i = blockIdx.x * blockDim.x + threadIdx.x;        // quad index
    int e = i << 2;                                       // element index
    if (e >= N_TOT * K_IN) return;
    int row = e >> 11;            // / 2048
    int col = e & (K_IN - 1);
    const float* src = (row < N_LAYER) ? (W1 + (size_t)row * K_IN + col)
                                       : (W2 + (size_t)(row - N_LAYER) * K_IN + col);
    v4f f = __builtin_nontemporal_load((const v4f* __restrict__)src);
    v4h h;
    h[0] = (_Float16)f[0];
    h[1] = (_Float16)f[1];
    h[2] = (_Float16)f[2];
    h[3] = (_Float16)f[3];
    *(v4h* __restrict__)(w16 + e) = h;
}

// ---------------------------------------------------------------------------
// WMMA GEMM: cur[t*B+b, n] = sum_k spikes_f16[t*B+b, k] * W_f16[n, k]
// Block tile 128(M) x 128(N); 8 waves in a 4x2 grid; each wave owns a
// 32x64 tile = 2x4 of v_wmma_f32_16x16x32_f16 (f32 accumulate).
// Per k-step per wave: 12 b128 loads feed 8 WMMAs (1.5 loads/WMMA); waves in
// the WGP share the WGP$, so cross-wave A/B re-loads are L0 hits.
// Raw currents go straight into the mem1/mem2 regions of d_out;
// the LIF kernel later overwrites them in place.
// ---------------------------------------------------------------------------
__global__ __launch_bounds__(256) void snn_gemm_wmma(const _Float16* __restrict__ sp,
                                                     const _Float16* __restrict__ w,
                                                     float* __restrict__ out) {
    const int lane = threadIdx.x & 31;
    const int wid  = threadIdx.x >> 5;       // 0..7
    const int wm   = wid >> 1;               // 0..3  -> row quadrant (32 rows)
    const int wn   = wid & 1;                // 0..1  -> col half (64 cols)
    const int lh   = lane & 15;              // lane within half-wave
    const int hi   = lane >> 4;              // half-wave select

    const int rowBase = blockIdx.y * 128 + wm * 32;   // over T*B = 16384
    const int colBase = blockIdx.x * 128 + wn * 64;   // over N_TOT = 2048

    // A: lane<16 -> row m, K {k0..k0+7, k0+16..k0+23}; lane>=16 -> +8 shift
    const _Float16* __restrict__ aRow0 = sp + (size_t)(rowBase + lh) * K_IN;
    const _Float16* __restrict__ aRow1 = aRow0 + (size_t)16 * K_IN;
    // B: lane n holds column n = W row n (contiguous); hi selects K halves
    const _Float16* __restrict__ bCol0 = w + (size_t)(colBase + lh) * K_IN;

    v8f c[2][4];
#pragma unroll
    for (int i = 0; i < 2; ++i)
#pragma unroll
        for (int j = 0; j < 4; ++j) c[i][j] = (v8f){};

    for (int k0 = 0; k0 < K_IN; k0 += 32) {
        // WGP-scope prefetch of the k-stream a few chunks ahead; one line per
        // lane covers only 16 rows per instruction (lanes l, l+16 hit the same row).
        prefetch_wgp(aRow0 + k0 + 128);
        prefetch_wgp(bCol0 + k0 + 128);

        FragAB a[2], b[4];
        const int ao = k0 + hi * 8;
#pragma unroll
        for (int i = 0; i < 2; ++i) {
            const _Float16* p = (i == 0) ? aRow0 : aRow1;
            a[i].h[0] = *(const v8h* __restrict__)(p + ao);
            a[i].h[1] = *(const v8h* __restrict__)(p + ao + 16);
        }
        const int bo = k0 + hi * 8;
#pragma unroll
        for (int j = 0; j < 4; ++j) {
            const _Float16* p = bCol0 + (size_t)(j * 16) * K_IN;  // imm-foldable
            b[j].h[0] = *(const v8h* __restrict__)(p + bo);
            b[j].h[1] = *(const v8h* __restrict__)(p + bo + 16);
        }
#pragma unroll
        for (int i = 0; i < 2; ++i)
#pragma unroll
            for (int j = 0; j < 4; ++j)
                c[i][j] = __builtin_amdgcn_wmma_f32_16x16x32_f16(
                    false, a[i].v, false, b[j].v, (short)0, c[i][j], false, false);
    }

    // Entire 128-wide block tile lies in one layer (1024 % 128 == 0)
    const int layer = colBase >> 10;                    // 0 -> W1, 1 -> W2
    float* __restrict__ memBase = out + (size_t)(2 * layer + 1) * TBN;
    // row r over T*B maps to [t][b] with t=r/256,b=r%256 -> flat r*1024
    const int col0 = (colBase & (N_LAYER - 1)) + lh;    // + subtile*16 below

#pragma unroll
    for (int i = 0; i < 2; ++i) {
#pragma unroll
        for (int j = 0; j < 4; ++j) {
#pragma unroll
            for (int jj = 0; jj < 8; ++jj) {
                const int r = rowBase + i * 16 + jj + 8 * hi;   // C/D layout
                const int cN = col0 + j * 16;
                memBase[(size_t)r * N_LAYER + cN] = c[i][j][jj];  // RT: re-read by LIF
            }
        }
    }
}

// ---------------------------------------------------------------------------
// LIF recursion (beta=0.9, thr=1.0, reset-by-subtraction). One thread per
// (b, n_global) element; reads raw current from the mem slot and overwrites
// it with the membrane value (same thread, same address -> safe), and writes
// spikes to the spk regions. Outputs are never re-read -> non-temporal stores
// so they stream to HBM without evicting the current buffer from L2.
// ---------------------------------------------------------------------------
__global__ __launch_bounds__(256) void lif_scan(float* __restrict__ out) {
    const int idx = blockIdx.x * blockDim.x + threadIdx.x;   // 0 .. B*N_TOT-1
    const int b   = idx >> 11;             // / 2048
    const int ng  = idx & (N_TOT - 1);
    const int layer = ng >> 10;
    const int n   = ng & (N_LAYER - 1);

    float* __restrict__ memP = out + (size_t)(2 * layer + 1) * TBN
                                   + (size_t)b * N_LAYER + n;
    float* __restrict__ spkP = out + (size_t)(2 * layer) * TBN
                                   + (size_t)b * N_LAYER + n;
    const size_t stride = (size_t)BATCH * N_LAYER;           // per-timestep

    float mem = 0.0f;
#pragma unroll 8
    for (int t = 0; t < T_STEPS; ++t) {
        const float cur = memP[(size_t)t * stride];
        const float reset = (mem - 1.0f > 0.0f) ? 1.0f : 0.0f;   // thr = 1
        mem = 0.9f * mem + cur - reset;                          // reset*thr
        const float spk = (mem - 1.0f > 0.0f) ? 1.0f : 0.0f;
        __builtin_nontemporal_store(mem, memP + (size_t)t * stride);
        __builtin_nontemporal_store(spk, spkP + (size_t)t * stride);
    }
}

// ---------------------------------------------------------------------------
extern "C" void kernel_launch(void* const* d_in, const int* in_sizes, int n_in,
                              void* d_out, int out_size, void* d_ws, size_t ws_size,
                              hipStream_t stream) {
    (void)in_sizes; (void)n_in; (void)out_size; (void)ws_size;
    const float* spikes = (const float*)d_in[0];   // [64,256,2048] f32 (0/1)
    const float* W1     = (const float*)d_in[1];   // [1024,2048] f32
    const float* W2     = (const float*)d_in[2];   // [1024,2048] f32
    float* out = (float*)d_out;                    // spk1|mem1|spk2|mem2

    _Float16* sp16 = (_Float16*)d_ws;                                  // 67 MB
    _Float16* w16  = sp16 + (size_t)T_STEPS * BATCH * K_IN;            // 8 MB

    // 1) precision conversion passes (L2-friendly streaming)
    const int spQuads = (T_STEPS * BATCH * K_IN) / 4;                  // 8388608
    cvt_spikes_f16<<<spQuads / 256, 256, 0, stream>>>(spikes, sp16, spQuads);
    const int wQuads = (N_TOT * K_IN) / 4;                             // 1048576
    pack_weights_f16<<<wQuads / 256, 256, 0, stream>>>(W1, W2, w16);

    // 2) WMMA GEMM for both layers (W1||W2), currents -> mem output regions
    dim3 grid(N_TOT / 128, (T_STEPS * BATCH) / 128);                   // 16 x 128
    snn_gemm_wmma<<<grid, 256, 0, stream>>>(sp16, w16, out);

    // 3) in-place LIF time recursion, emits spk + mem
    lif_scan<<<(BATCH * N_TOT) / 256, 256, 0, stream>>>(out);
}
